// CrystalGraphConvNet_27341761806838
// MI455X (gfx1250) — matrix operand order repaired and checked by reading, not compiled, not run on hardware
//
#include <hip/hip_runtime.h>
#include <hip/hip_bf16.h>

// ---------------------------------------------------------------------------
// CGCNN forward for MI455X (gfx1250, wave32, WMMA).
// Dominant GEMM (1.2M x 169 x 128, x3 layers) runs on v_wmma_f32_16x16x32_bf16.
// Workspace use: ~360 MB (x 25.6MB, z bf16 307MB, summed 25.6MB, small tails).
// ---------------------------------------------------------------------------

typedef __attribute__((ext_vector_type(16))) __bf16 v16bf;
typedef __attribute__((ext_vector_type(8)))  float  v8f;

#define N_ATOMS   100000
#define M_NBR     12
#define F_ATOM    64            // ATOM_F
#define NF        128           // 2*ATOM_F (GEMM output cols / BN1 channels)
#define NBR_F     41
#define KTOT      169           // 2*F + 41
#define KP        192           // K padded to 6 chunks of 32 (bf16 WMMA K)
#define N_CRYST   2500
#define ORIG_F    92
#define NROWS     (N_ATOMS * M_NBR)     // 1,200,000
#define NTILES    (NROWS / 16)          // 75,000 (exact)
#define WPB       8                     // waves per block
#define GEMM_BLOCKS (NTILES / WPB)      // 9,375 (exact)

__device__ __forceinline__ float softplusf(float v) {
    return (v > 20.0f) ? v : log1pf(__expf(v));
}

// x = atom_fea @ emb_W + emb_b  : (100000,92)@(92,64). Tiny; VALU kernel.
// grid covers N_ATOMS*F_ATOM exactly.
__global__ void embed_kernel(const float* __restrict__ af,
                             const float* __restrict__ W,
                             const float* __restrict__ b,
                             float* __restrict__ x)
{
    const int i = blockIdx.x * blockDim.x + threadIdx.x;
    const int n = i >> 6;
    const int f = i & 63;
    float s = b[f];
    #pragma unroll 4
    for (int k = 0; k < ORIG_F; ++k)
        s = fmaf(af[n * ORIG_F + k], W[k * F_ATOM + f], s);
    x[i] = s;
}

// conv_W (169x128 f32, k-major) -> Wt (128x192 bf16, n-major, zero-padded K).
__global__ void prep_wt(const float* __restrict__ W, __bf16* __restrict__ Wt)
{
    const int i = blockIdx.x * blockDim.x + threadIdx.x;
    if (i >= NF * KP) return;
    const int n = i / KP;
    const int k = i - n * KP;
    const float v = (k < KTOT) ? W[k * NF + n] : 0.0f;
    Wt[i] = (__bf16)v;
}

// Core GEMM: per wave, one 16-row x 128-col tile of
//   z = [x_self | x_nbr | nbr_fea] @ W + bias
// using v_wmma_f32_16x16x32_bf16, K = 6 chunks of 32.
// Also accumulates per-column sum / sum-of-squares for BN1 (LDS reduce, then
// one global float atomic per column per block).
__global__ void __launch_bounds__(256)
conv_gemm_wmma(const float* __restrict__ x,
               const float* __restrict__ nbr_fea,
               const int*   __restrict__ nbr_idx,
               const __bf16* __restrict__ Wt,     // [128][192] bf16
               const float* __restrict__ bias,    // [128]
               __bf16* __restrict__ z,            // [NROWS][128] bf16
               float* __restrict__ stats)         // [256]: sum[128], sumsq[128]
{
    __shared__ __align__(32) __bf16 a_lds[WPB][16 * KP];  // 8 * 6KB = 48KB
    __shared__ float s_stats[2 * NF];

    const int lane = threadIdx.x & 31;
    const int wave = threadIdx.x >> 5;

    for (int i = threadIdx.x; i < 2 * NF; i += blockDim.x) s_stats[i] = 0.0f;

    const int tile = blockIdx.x * WPB + wave;   // always < NTILES (exact grid)
    const int row0 = tile * 16;
    __bf16* A = a_lds[wave];

    // Stage the 16x192 A tile (bf16, zero-padded) cooperatively within the wave.
    for (int i = lane; i < 16 * KP; i += 32) {
        const int r  = i / KP;
        const int c  = i - r * KP;
        const int gr = row0 + r;                // global (atom,neighbor) row
        const int n  = gr / M_NBR;
        float v;
        if (c < F_ATOM)            v = x[n * F_ATOM + c];
        else if (c < 2 * F_ATOM)   v = x[nbr_idx[gr] * F_ATOM + (c - F_ATOM)];
        else if (c < KTOT)         v = nbr_fea[gr * NBR_F + (c - 2 * F_ATOM)];
        else                       v = 0.0f;
        A[r * KP + c] = (__bf16)v;
    }
    __syncthreads();   // uniform: grid exactly covers all tiles

    const int h  = lane >> 4;     // ISA lane-half (K interleave)
    const int mn = lane & 15;     // A: row M; B/D: col N

    v8f acc[8] = {};
    #pragma unroll
    for (int c = 0; c < 6; ++c) {
        // A fragment, contiguous K order matching B: elem e -> K=32c+16h+e.
        // Two aligned 16B LDS loads.
        union { v16bf v; uint4 q[2]; } af;
        const __bf16* ap = A + mn * KP + 32 * c + 16 * h;
        af.q[0] = *(const uint4*)(ap);
        af.q[1] = *(const uint4*)(ap + 8);
        #pragma unroll
        for (int nc = 0; nc < 8; ++nc) {
            // B fragment: lane holds Wt[N][32c+16h .. +15], one aligned 32B load
            const __bf16* bp = Wt + (nc * 16 + mn) * KP + 32 * c + 16 * h;
            const v16bf bf = *(const v16bf*)bp;
            acc[nc] = __builtin_amdgcn_wmma_f32_16x16x32_bf16(
                false, af.v, false, bf, (short)0, acc[nc], false, false);
        }
    }

    // Bias add, bf16 z writeback (D layout: elem r -> row 8h+r, col 16nc+mn),
    // per-column BN1 partial stats into LDS.
    #pragma unroll
    for (int nc = 0; nc < 8; ++nc) {
        const int col = nc * 16 + mn;
        const float bc = bias[col];
        float s = 0.0f, s2 = 0.0f;
        #pragma unroll
        for (int r = 0; r < 8; ++r) {
            const float v = acc[nc][r] + bc;
            const int grow = row0 + 8 * h + r;
            z[(long long)grow * NF + col] = (__bf16)v;
            s  += v;
            s2 += v * v;
        }
        atomicAdd(&s_stats[col], s);
        atomicAdd(&s_stats[NF + col], s2);
    }
    __syncthreads();
    for (int i = threadIdx.x; i < 2 * NF; i += blockDim.x)
        atomicAdd(&stats[i], s_stats[i]);
}

// Turn (sum, sumsq, gamma, beta) into per-channel affine scale/shift.
__global__ void bn_finalize(const float* __restrict__ stats,
                            const float* __restrict__ g,
                            const float* __restrict__ b,
                            float* __restrict__ scale,
                            float* __restrict__ shift,
                            int C, float invN)
{
    const int c = threadIdx.x;
    if (c < C) {
        const float mean = stats[c] * invN;
        const float var  = stats[C + c] * invN - mean * mean;
        const float sc   = g[c] * rsqrtf(var + 1e-5f);
        scale[c] = sc;
        shift[c] = b[c] - mean * sc;
    }
}

// summed[n,f] = sum_m sigmoid(bn1(filt)) * softplus(bn1(core)); BN2 stats too.
// grid covers N_ATOMS*F_ATOM exactly.
__global__ void __launch_bounds__(256)
pool_kernel(const __bf16* __restrict__ z,
            const float* __restrict__ scale1,
            const float* __restrict__ shift1,
            float* __restrict__ summed,
            float* __restrict__ stats2)     // [128]: sum[64], sumsq[64]
{
    __shared__ float s_stats[2 * F_ATOM];
    for (int i = threadIdx.x; i < 2 * F_ATOM; i += blockDim.x) s_stats[i] = 0.0f;
    __syncthreads();

    const int i = blockIdx.x * blockDim.x + threadIdx.x;
    const int n = i >> 6;
    const int f = i & 63;
    const float scF = scale1[f],          shF = shift1[f];
    const float scC = scale1[F_ATOM + f], shC = shift1[F_ATOM + f];
    float s = 0.0f;
    const __bf16* zr = z + (long long)n * M_NBR * NF;
    #pragma unroll
    for (int m = 0; m < M_NBR; ++m, zr += NF) {
        const float filt = (float)zr[f]          * scF + shF;
        const float core = (float)zr[F_ATOM + f] * scC + shC;
        const float sig  = 1.0f / (1.0f + __expf(-filt));
        s += sig * softplusf(core);
    }
    summed[i] = s;
    atomicAdd(&s_stats[f], s);
    atomicAdd(&s_stats[F_ATOM + f], s * s);
    __syncthreads();
    for (int t = threadIdx.x; t < 2 * F_ATOM; t += blockDim.x)
        atomicAdd(&stats2[t], s_stats[t]);
}

// x = softplus(x + bn2(summed))     (in place; grid exact)
__global__ void update_kernel(float* __restrict__ x,
                              const float* __restrict__ summed,
                              const float* __restrict__ scale2,
                              const float* __restrict__ shift2)
{
    const int i = blockIdx.x * blockDim.x + threadIdx.x;
    const int f = i & 63;
    const float v = x[i] + summed[i] * scale2[f] + shift2[f];
    x[i] = softplusf(v);
}

// Segment sum + counts over crystals (grid exact).
__global__ void segment_kernel(const float* __restrict__ x,
                               const int* __restrict__ cid,
                               float* __restrict__ sums,
                               float* __restrict__ counts)
{
    const int i = blockIdx.x * blockDim.x + threadIdx.x;
    const int n = i >> 6;
    const int f = i & 63;
    const int c = cid[n];
    atomicAdd(&sums[c * F_ATOM + f], x[i]);
    if (f == 0) atomicAdd(&counts[c], 1.0f);
}

// Per-crystal head: mean -> softplus(fc) -> out.  One block per crystal.
// d_out layout: out[2500] first, then crys_fea[2500*128] (tuple return order).
__global__ void head_kernel(const float* __restrict__ sums,
                            const float* __restrict__ counts,
                            const float* __restrict__ fcW,   // [64][128]
                            const float* __restrict__ fcb,   // [128]
                            const float* __restrict__ outW,  // [128]
                            const float* __restrict__ outb,  // [1]
                            float* __restrict__ d_out)
{
    __shared__ float crys[F_ATOM];
    __shared__ float red[128];
    const int c = blockIdx.x;
    const int j = threadIdx.x;
    const float cnt = fmaxf(counts[c], 1.0f);
    if (j < F_ATOM) crys[j] = sums[c * F_ATOM + j] / cnt;
    __syncthreads();
    float s = fcb[j];
    #pragma unroll 8
    for (int k = 0; k < F_ATOM; ++k)
        s = fmaf(crys[k], fcW[k * 128 + j], s);
    const float h = softplusf(s);
    d_out[N_CRYST + c * 128 + j] = h;          // crys_fea
    red[j] = h * outW[j];
    __syncthreads();
    for (int off = 64; off > 0; off >>= 1) {
        if (j < off) red[j] += red[j + off];
        __syncthreads();
    }
    if (j == 0) d_out[c] = red[0] + outb[0];   // out
}

extern "C" void kernel_launch(void* const* d_in, const int* in_sizes, int n_in,
                              void* d_out, int out_size, void* d_ws, size_t ws_size,
                              hipStream_t stream) {
    (void)in_sizes; (void)n_in; (void)out_size; (void)ws_size;

    const float* atom_fea = (const float*)d_in[0];
    const float* nbr_fea  = (const float*)d_in[1];
    const int*   nbr_idx  = (const int*)d_in[2];
    const int*   cid      = (const int*)d_in[3];
    const float* emb_W    = (const float*)d_in[4];
    const float* emb_b    = (const float*)d_in[5];
    const float* conv_W   = (const float*)d_in[6];   // [3][169][128]
    const float* conv_b   = (const float*)d_in[7];   // [3][128]
    const float* bn1_g    = (const float*)d_in[8];   // [3][128]
    const float* bn1_b    = (const float*)d_in[9];
    const float* bn2_g    = (const float*)d_in[10];  // [3][64]
    const float* bn2_b    = (const float*)d_in[11];
    const float* fc_W     = (const float*)d_in[12];  // [64][128]
    const float* fc_b     = (const float*)d_in[13];
    const float* out_W    = (const float*)d_in[14];  // [128][1]
    const float* out_b    = (const float*)d_in[15];
    float* out = (float*)d_out;

    // Workspace partition (all 256B aligned).
    char* p = (char*)d_ws;
    auto take = [&](size_t bytes) {
        char* r = p;
        p += (bytes + 255) & ~(size_t)255;
        return r;
    };
    float*  x      = (float*) take(sizeof(float)  * (size_t)N_ATOMS * F_ATOM);
    __bf16* z      = (__bf16*)take(sizeof(__bf16) * (size_t)NROWS * NF);
    __bf16* Wt     = (__bf16*)take(sizeof(__bf16) * NF * KP);
    float*  stats1 = (float*) take(sizeof(float)  * 2 * NF);
    float*  scale1 = (float*) take(sizeof(float)  * NF);
    float*  shift1 = (float*) take(sizeof(float)  * NF);
    float*  stats2 = (float*) take(sizeof(float)  * 2 * F_ATOM);
    float*  scale2 = (float*) take(sizeof(float)  * F_ATOM);
    float*  shift2 = (float*) take(sizeof(float)  * F_ATOM);
    float*  summed = (float*) take(sizeof(float)  * (size_t)N_ATOMS * F_ATOM);
    float*  csums  = (float*) take(sizeof(float)  * (N_CRYST * F_ATOM + N_CRYST));
    float*  counts = csums + N_CRYST * F_ATOM;

    const int NXF_BLOCKS = (N_ATOMS * F_ATOM) / 256;   // 25000, exact

    embed_kernel<<<NXF_BLOCKS, 256, 0, stream>>>(atom_fea, emb_W, emb_b, x);

    for (int L = 0; L < 3; ++L) {
        prep_wt<<<(NF * KP + 255) / 256, 256, 0, stream>>>(conv_W + (size_t)L * KTOT * NF, Wt);
        hipMemsetAsync(stats1, 0, sizeof(float) * 2 * NF, stream);
        conv_gemm_wmma<<<GEMM_BLOCKS, 256, 0, stream>>>(
            x, nbr_fea, nbr_idx, Wt, conv_b + L * NF, z, stats1);
        bn_finalize<<<1, NF, 0, stream>>>(stats1, bn1_g + L * NF, bn1_b + L * NF,
                                          scale1, shift1, NF, 1.0f / (float)NROWS);
        hipMemsetAsync(stats2, 0, sizeof(float) * 2 * F_ATOM, stream);
        pool_kernel<<<NXF_BLOCKS, 256, 0, stream>>>(z, scale1, shift1, summed, stats2);
        bn_finalize<<<1, F_ATOM, 0, stream>>>(stats2, bn2_g + L * F_ATOM, bn2_b + L * F_ATOM,
                                              scale2, shift2, F_ATOM, 1.0f / (float)N_ATOMS);
        update_kernel<<<NXF_BLOCKS, 256, 0, stream>>>(x, summed, scale2, shift2);
    }

    hipMemsetAsync(csums, 0, sizeof(float) * (N_CRYST * F_ATOM + N_CRYST), stream);
    segment_kernel<<<NXF_BLOCKS, 256, 0, stream>>>(x, cid, csums, counts);
    head_kernel<<<N_CRYST, 128, 0, stream>>>(csums, counts, fc_W, fc_b, out_W, out_b, out);
}
